// Informer_27839978012797
// MI455X (gfx1250) — matrix operand order repaired and checked by reading, not compiled, not run on hardware
//
#include <hip/hip_runtime.h>
#include <hip/hip_bf16.h>
#include <math.h>

#define BB 16
#define HH 8
#define DH 64
#define DMODEL 512

typedef _Float16 f16x16 __attribute__((ext_vector_type(16)));
typedef _Float16 f16x8  __attribute__((ext_vector_type(8)));
typedef _Float16 f16x4  __attribute__((ext_vector_type(4)));
typedef float    f32x8  __attribute__((ext_vector_type(8)));
typedef float    f32x4  __attribute__((ext_vector_type(4)));

__device__ __forceinline__ unsigned hash_u32(unsigned x) {
  x ^= x >> 16; x *= 0x7feb352du;
  x ^= x >> 15; x *= 0x846ca68bu;
  x ^= x >> 16; return x;
}

// ---------------------------------------------------------------------------
// WMMA GEMM: C[M,N] = act(A'[M,K] @ B[K,N] + bias), optional += accumulate.
// A' row m reads A row (batch-local circular shift): b = m / Lseg, l = m % Lseg,
// src = b*Lseg + (l + shift mod Lseg).  f32 inputs converted to f16 in LDS,
// f32 accumulate via v_wmma_f32_16x16x32_f16.
// Tile: 128x128 per block (256 thr = 8 waves), each wave 32x64 (2x4 WMMA).
// ---------------------------------------------------------------------------
__global__ __launch_bounds__(256)
void k_wmma_gemm(const float* __restrict__ A, const float* __restrict__ Bw,
                 const float* __restrict__ bias, float* __restrict__ C,
                 int M, int N, int K, int act, int accum, int shift, int Lseg)
{
  __shared__ _Float16 sA[128][48];   // [m][k], row stride 96B (32B aligned)
  __shared__ _Float16 sB[128][48];   // [n][k] (transposed B tile)

  const int tid  = threadIdx.x;
  const int wave = tid >> 5;
  const int lane = tid & 31;
  const int l16  = lane & 15;
  const int kh   = lane >> 4;        // lane group 0/1
  const int wm   = wave >> 1;        // 0..3
  const int wn   = wave & 1;         // 0..1
  const int m0   = blockIdx.y * 128;
  const int n0   = blockIdx.x * 128;

  f32x8 acc[2][4];
  for (int i = 0; i < 2; ++i)
    for (int j = 0; j < 4; ++j)
      for (int e = 0; e < 8; ++e) acc[i][j][e] = 0.0f;

  for (int k0 = 0; k0 < K; k0 += 32) {
    // ---- stage A tile 128x32 (f32 -> f16) ----
    for (int i = tid; i < 1024; i += 256) {
      int r  = i >> 3;         // 0..127
      int c4 = i & 7;          // float4 index within 32-wide K slab
      int mg = m0 + r;
      f32x4 av;
      if (mg < M && (k0 + c4 * 4) < K) {
        int bsel = mg / Lseg;
        int l    = mg - bsel * Lseg;
        int ls   = l + shift;
        if (ls < 0) ls += Lseg; else if (ls >= Lseg) ls -= Lseg;
        const float* src = A + (size_t)(bsel * Lseg + ls) * K + k0 + c4 * 4;
        av = *(const f32x4*)src;
        if (k0 + 32 < K) __builtin_prefetch(src + 32, 0, 0);  // global_prefetch_b8
      } else {
        av[0] = av[1] = av[2] = av[3] = 0.0f;
      }
      f16x4 h;
      h[0] = (_Float16)av[0]; h[1] = (_Float16)av[1];
      h[2] = (_Float16)av[2]; h[3] = (_Float16)av[3];
      *(f16x4*)&sA[r][c4 * 4] = h;
    }
    // ---- stage B tile 32x128, transposed to [n][k] ----
    for (int i = tid; i < 1024; i += 256) {
      int kk = i >> 5;         // 0..31
      int c4 = i & 31;         // 0..31 -> cols c4*4..+3
      int ng = n0 + c4 * 4;
      f32x4 bv;
      if ((k0 + kk) < K && ng < N) {
        bv = *(const f32x4*)(Bw + (size_t)(k0 + kk) * N + ng);
      } else {
        bv[0] = bv[1] = bv[2] = bv[3] = 0.0f;
      }
      for (int j = 0; j < 4; ++j) sB[c4 * 4 + j][kk] = (_Float16)bv[j];
    }
    __syncthreads();

    // ---- fragments + WMMA ----
    f16x16 af[2], bf[4];
    for (int i = 0; i < 2; ++i) {
      int row = wm * 32 + i * 16 + l16;
      f16x8 lo = *(const f16x8*)&sA[row][8 * kh];
      f16x8 hi = *(const f16x8*)&sA[row][16 + 8 * kh];
      af[i] = __builtin_shufflevector(lo, hi, 0,1,2,3,4,5,6,7,8,9,10,11,12,13,14,15);
    }
    for (int j = 0; j < 4; ++j) {
      int col = wn * 64 + j * 16 + l16;
      f16x8 blo = *(const f16x8*)&sB[col][8 * kh];
      f16x8 bhi = *(const f16x8*)&sB[col][16 + 8 * kh];
      bf[j] = __builtin_shufflevector(blo, bhi, 0,1,2,3,4,5,6,7,8,9,10,11,12,13,14,15);
    }
    for (int i = 0; i < 2; ++i)
      for (int j = 0; j < 4; ++j)
        acc[i][j] = __builtin_amdgcn_wmma_f32_16x16x32_f16(
            false, af[i], false, bf[j], (short)0, acc[i][j], false, false);
    __syncthreads();
  }

  // ---- epilogue: bias, GELU(exact), store / accumulate ----
  for (int i = 0; i < 2; ++i)
    for (int j = 0; j < 4; ++j) {
      int nl = n0 + wn * 64 + j * 16 + l16;
      if (nl >= N) continue;
      float bs = bias ? bias[nl] : 0.0f;
      for (int r = 0; r < 8; ++r) {
        int ml = m0 + wm * 32 + i * 16 + r + kh * 8;
        if (ml >= M) continue;
        float v = acc[i][j][r] + bs;
        if (act == 1) v = 0.5f * v * (1.0f + erff(v * 0.70710678118f));
        size_t idx = (size_t)ml * N + nl;
        if (accum) C[idx] += v; else C[idx] = v;
      }
    }
}

// ---------------------------------------------------------------------------
// Data embedding: circular conv1d(C_IN=16 -> 512, k=3) + positional + temporal
// sinusoids.  seq: (B, 16, L), mark: (B, L, 4) int32, out: (B, L, 512).
// ---------------------------------------------------------------------------
__global__ void k_embed(const float* __restrict__ seq, const int* __restrict__ mark,
                        const float* __restrict__ cw, const float* __restrict__ cb,
                        float* __restrict__ out, int L)
{
  size_t t = (size_t)blockIdx.x * blockDim.x + threadIdx.x;
  size_t total = (size_t)BB * L * DMODEL;
  if (t >= total) return;
  int d = (int)(t % DMODEL);
  int l = (int)((t / DMODEL) % L);
  int b = (int)(t / ((size_t)DMODEL * L));

  int lm = (l + L - 1) % L;
  int lp = (l + 1) % L;
  float a = cb[d];
  for (int c = 0; c < 16; ++c) {
    const float* s  = seq + ((size_t)b * 16 + c) * L;
    const float* wr = cw + ((size_t)d * 16 + c) * 3;
    a += s[lm] * wr[0] + s[l] * wr[1] + s[lp] * wr[2];
  }
  int j = d >> 1;
  float freq = __expf(-(float)(2 * j) * (9.210340371976184f / 512.0f));
  float pe = (d & 1) ? __cosf((float)l * freq) : __sinf((float)l * freq);
  const int* mr = mark + ((size_t)b * L + l) * 4;
  float te = 0.0f;
  for (int m = 0; m < 4; ++m) {
    float ang = (float)mr[m] * freq;
    te += (d & 1) ? __cosf(ang) : __sinf(ang);
  }
  out[t] = a + pe + te;
}

// ---------- ProbSparse attention pieces ----------
// q,k,v layouts: (B, L, H*DH) row-major.  ctx layout: (B, H, Lq, DH).
__global__ void k_sampleM(const float* __restrict__ q, const float* __restrict__ k,
                          float* __restrict__ Mout, int Lq, int Lk, int U, unsigned seed)
{
  int t = blockIdx.x * blockDim.x + threadIdx.x;
  int total = BB * HH * Lq;
  if (t >= total) return;
  int l = t % Lq; int h = (t / Lq) % HH; int b = t / (Lq * HH);
  const float* qr = q + ((size_t)(b * Lq + l)) * DMODEL + h * DH;
  float mx = -INFINITY, sm = 0.0f;
  for (int s = 0; s < U; ++s) {
    unsigned r = hash_u32(seed ^ ((unsigned)(l * U + s) * 2654435761u));
    int kk = (int)(r % (unsigned)Lk);
    const float* kr = k + ((size_t)(b * Lk + kk)) * DMODEL + h * DH;
    float dot = 0.0f;
    for (int d = 0; d < DH; ++d) dot += qr[d] * kr[d];
    mx = fmaxf(mx, dot); sm += dot;
  }
  Mout[t] = mx - sm / (float)U;
}

__global__ void k_topk(const float* __restrict__ Min, int* __restrict__ top, int Lq, int u)
{
  __shared__ float vals[2048];
  __shared__ float rv[256];
  __shared__ int   ri[256];
  int bh = blockIdx.x, tid = threadIdx.x;
  const float* m = Min + (size_t)bh * Lq;
  for (int i = tid; i < Lq; i += 256) vals[i] = m[i];
  __syncthreads();
  for (int it = 0; it < u; ++it) {
    float best = -INFINITY; int bi = 0;
    for (int i = tid; i < Lq; i += 256) {
      float v = vals[i];
      if (v > best) { best = v; bi = i; }
    }
    rv[tid] = best; ri[tid] = bi;
    __syncthreads();
    for (int s = 128; s > 0; s >>= 1) {
      if (tid < s && rv[tid + s] > rv[tid]) { rv[tid] = rv[tid + s]; ri[tid] = ri[tid + s]; }
      __syncthreads();
    }
    if (tid == 0) { top[(size_t)bh * u + it] = ri[0]; vals[ri[0]] = -INFINITY; }
    __syncthreads();
  }
}

__global__ void k_scores(const float* __restrict__ q, const float* __restrict__ k,
                         const int* __restrict__ top, float* __restrict__ S,
                         int Lq, int Lk, int u, int causal)
{
  size_t t = (size_t)blockIdx.x * blockDim.x + threadIdx.x;
  size_t total = (size_t)BB * HH * u * Lk;
  if (t >= total) return;
  int kk = (int)(t % Lk);
  int ui = (int)((t / Lk) % u);
  int h  = (int)((t / ((size_t)Lk * u)) % HH);
  int b  = (int)(t / ((size_t)Lk * u * HH));
  int l  = top[(size_t)(b * HH + h) * u + ui];
  const float* qr = q + ((size_t)(b * Lq + l)) * DMODEL + h * DH;
  const float* kr = k + ((size_t)(b * Lk + kk)) * DMODEL + h * DH;
  float dot = 0.0f;
  for (int d = 0; d < DH; ++d) dot += qr[d] * kr[d];
  dot *= 0.125f;  // 1/sqrt(64)
  if (causal && kk > l) dot = -INFINITY;
  S[t] = dot;
}

__global__ void k_softmax(float* __restrict__ S, int Lk)
{
  __shared__ float red[256];
  int tid = threadIdx.x;
  float* row = S + (size_t)blockIdx.x * Lk;
  float mx = -INFINITY;
  for (int i = tid; i < Lk; i += 256) mx = fmaxf(mx, row[i]);
  red[tid] = mx; __syncthreads();
  for (int s = 128; s > 0; s >>= 1) { if (tid < s) red[tid] = fmaxf(red[tid], red[tid + s]); __syncthreads(); }
  mx = red[0]; __syncthreads();
  float sm = 0.0f;
  for (int i = tid; i < Lk; i += 256) { float e = __expf(row[i] - mx); row[i] = e; sm += e; }
  red[tid] = sm; __syncthreads();
  for (int s = 128; s > 0; s >>= 1) { if (tid < s) red[tid] += red[tid + s]; __syncthreads(); }
  float inv = 1.0f / red[0];
  for (int i = tid; i < Lk; i += 256) row[i] *= inv;
}

__global__ void k_vmean(const float* __restrict__ v, float* __restrict__ vm, int Lk)
{
  int t = blockIdx.x * blockDim.x + threadIdx.x;
  if (t >= BB * HH * DH) return;
  int d = t % DH; int h = (t / DH) % HH; int b = t / (DH * HH);
  float s = 0.0f;
  for (int l = 0; l < Lk; ++l) s += v[((size_t)(b * Lk + l)) * DMODEL + h * DH + d];
  vm[t] = s / (float)Lk;
}

__global__ void k_ctx_fill(const float* __restrict__ vm, float* __restrict__ ctx, int Lq)
{
  size_t t = (size_t)blockIdx.x * blockDim.x + threadIdx.x;
  size_t total = (size_t)BB * HH * Lq * DH;
  if (t >= total) return;
  int d = (int)(t % DH);
  int h = (int)((t / ((size_t)DH * Lq)) % HH);
  int b = (int)(t / ((size_t)DH * Lq * HH));
  ctx[t] = vm[(b * HH + h) * DH + d];
}

__global__ void k_ctx_cumsum(const float* __restrict__ v, float* __restrict__ ctx, int Lq)
{
  int t = blockIdx.x * blockDim.x + threadIdx.x;
  if (t >= BB * HH * DH) return;
  int d = t % DH; int h = (t / DH) % HH; int b = t / (DH * HH);
  float s = 0.0f;
  for (int l = 0; l < Lq; ++l) {
    s += v[((size_t)(b * Lq + l)) * DMODEL + h * DH + d];
    ctx[((size_t)((b * HH + h) * Lq + l)) * DH + d] = s;
  }
}

__global__ void k_upd_scatter(const float* __restrict__ S, const float* __restrict__ v,
                              const int* __restrict__ top, float* __restrict__ ctx,
                              int Lq, int Lk, int u)
{
  int t = blockIdx.x * blockDim.x + threadIdx.x;
  int total = BB * HH * u * DH;
  if (t >= total) return;
  int d  = t % DH;
  int ui = (t / DH) % u;
  int h  = (t / (DH * u)) % HH;
  int b  = t / (DH * u * HH);
  const float* srow = S + (size_t)((b * HH + h) * u + ui) * Lk;
  float s = 0.0f;
  for (int kk = 0; kk < Lk; ++kk)
    s += srow[kk] * v[((size_t)(b * Lk + kk)) * DMODEL + h * DH + d];
  int l = top[(size_t)(b * HH + h) * u + ui];
  ctx[((size_t)((b * HH + h) * Lq + l)) * DH + d] = s;
}

// ctx (B,H,L,DH) -> out (B,L,H*DH)   (mix=False layout)
__global__ void k_mixfalse(const float* __restrict__ ctx, float* __restrict__ out, int Lq)
{
  size_t t = (size_t)blockIdx.x * blockDim.x + threadIdx.x;
  size_t total = (size_t)BB * HH * Lq * DH;
  if (t >= total) return;
  int d = (int)(t % DH);
  int l = (int)((t / DH) % Lq);
  int h = (int)((t / ((size_t)DH * Lq)) % HH);
  int b = (int)(t / ((size_t)DH * Lq * HH));
  out[((size_t)(b * Lq + l)) * DMODEL + h * DH + d] = ctx[t];
}

// ---------- fused full (cross) attention: block per (b,h,lq), 128 threads ----------
__global__ void k_fullattn(const float* __restrict__ q, const float* __restrict__ k,
                           const float* __restrict__ v, float* __restrict__ out,
                           int Lq, int Lk)
{
  __shared__ float p[512];
  __shared__ float red[128];
  int lq = blockIdx.x % Lq;
  int h  = (blockIdx.x / Lq) % HH;
  int b  = blockIdx.x / (Lq * HH);
  int tid = threadIdx.x;
  const float* qr = q + ((size_t)(b * Lq + lq)) * DMODEL + h * DH;

  float mx = -INFINITY;
  for (int kk = tid; kk < Lk; kk += 128) {
    const float* kr = k + ((size_t)(b * Lk + kk)) * DMODEL + h * DH;
    float dot = 0.0f;
    for (int d = 0; d < DH; ++d) dot += qr[d] * kr[d];
    dot *= 0.125f;
    p[kk] = dot;
    mx = fmaxf(mx, dot);
  }
  red[tid] = mx; __syncthreads();
  for (int s = 64; s > 0; s >>= 1) { if (tid < s) red[tid] = fmaxf(red[tid], red[tid + s]); __syncthreads(); }
  mx = red[0]; __syncthreads();
  float sm = 0.0f;
  for (int kk = tid; kk < Lk; kk += 128) { float e = __expf(p[kk] - mx); p[kk] = e; sm += e; }
  red[tid] = sm; __syncthreads();
  for (int s = 64; s > 0; s >>= 1) { if (tid < s) red[tid] += red[tid + s]; __syncthreads(); }
  float inv = 1.0f / red[0];
  __syncthreads();

  int d = tid & 63;
  int half = tid >> 6;
  int span = Lk / 2;
  float a = 0.0f;
  for (int kk = half * span; kk < (half + 1) * span; ++kk)
    a += p[kk] * v[((size_t)(b * Lk + kk)) * DMODEL + h * DH + d];
  red[tid] = a; __syncthreads();
  if (half == 0)
    out[((size_t)(b * Lq + lq)) * DMODEL + h * DH + d] = (red[tid] + red[tid + 64]) * inv;
}

// ---------- residual + layernorm over D=512 (y nullable) ----------
__global__ void k_add_ln(const float* __restrict__ x, const float* __restrict__ y,
                         const float* __restrict__ g, const float* __restrict__ b,
                         float* __restrict__ out)
{
  __shared__ float r1[256], r2[256];
  int tid = threadIdx.x;
  size_t row = blockIdx.x;
  const float* xr = x + row * DMODEL;
  float v0 = xr[tid], v1 = xr[tid + 256];
  if (y) { const float* yr = y + row * DMODEL; v0 += yr[tid]; v1 += yr[tid + 256]; }
  r1[tid] = v0 + v1; r2[tid] = v0 * v0 + v1 * v1;
  __syncthreads();
  for (int s = 128; s > 0; s >>= 1) {
    if (tid < s) { r1[tid] += r1[tid + s]; r2[tid] += r2[tid + s]; }
    __syncthreads();
  }
  float mean = r1[0] * (1.0f / 512.0f);
  float var  = r2[0] * (1.0f / 512.0f) - mean * mean;
  float inv  = rsqrtf(var + 1e-5f);
  float* orow = out + row * DMODEL;
  orow[tid]       = (v0 - mean) * inv * g[tid]       + b[tid];
  orow[tid + 256] = (v1 - mean) * inv * g[tid + 256] + b[tid + 256];
}

// ---------- distilling conv helpers ----------
__global__ void k_packw(const float* __restrict__ w, float* __restrict__ pw)
{
  int t = blockIdx.x * blockDim.x + threadIdx.x;
  if (t >= 3 * DMODEL * DMODEL) return;
  int n = t % DMODEL;
  int kk = (t / DMODEL) % DMODEL;
  int tt = t / (DMODEL * DMODEL);
  pw[t] = w[((size_t)n * DMODEL + kk) * 3 + tt];   // w: (O, I, 3)
}

__global__ void k_bnstats(const float* __restrict__ y, float* __restrict__ mean,
                          float* __restrict__ var, int M)
{
  __shared__ float r1[256], r2[256];
  int d = blockIdx.x, tid = threadIdx.x;
  float s = 0.0f, s2 = 0.0f;
  for (int i = tid; i < M; i += 256) {
    float v = y[(size_t)i * DMODEL + d];
    s += v; s2 += v * v;
  }
  r1[tid] = s; r2[tid] = s2;
  __syncthreads();
  for (int st = 128; st > 0; st >>= 1) {
    if (tid < st) { r1[tid] += r1[tid + st]; r2[tid] += r2[tid + st]; }
    __syncthreads();
  }
  if (tid == 0) {
    float m = r1[0] / (float)M;
    mean[d] = m;
    var[d]  = r2[0] / (float)M - m * m;
  }
}

__global__ void k_bnpool(const float* __restrict__ y, const float* __restrict__ mean,
                         const float* __restrict__ var, const float* __restrict__ gg,
                         const float* __restrict__ bb, float* __restrict__ out,
                         int L, int Lo)
{
  size_t t = (size_t)blockIdx.x * blockDim.x + threadIdx.x;
  size_t total = (size_t)BB * Lo * DMODEL;
  if (t >= total) return;
  int d  = (int)(t % DMODEL);
  int lo = (int)((t / DMODEL) % Lo);
  int b  = (int)(t / ((size_t)DMODEL * Lo));
  float mu = mean[d], iv = rsqrtf(var[d] + 1e-5f), ga = gg[d], be = bb[d];
  float best = -INFINITY;
  for (int tt = 0; tt < 3; ++tt) {
    int li = 2 * lo - 1 + tt;
    if (li < 0 || li >= L) continue;
    float yv = (y[((size_t)b * L + li) * DMODEL + d] - mu) * iv * ga + be;
    float e = yv > 0.0f ? yv : (__expf(yv) - 1.0f);   // ELU
    best = fmaxf(best, e);
  }
  out[t] = best;
}

// ---------- final projections ----------
__global__ void k_lastproj(const float* __restrict__ dec, const float* __restrict__ pw,
                           const float* __restrict__ pb, float* __restrict__ last16)
{
  int tid = threadIdx.x;             // 256 threads = 16*16
  int b = tid / 16, c = tid % 16;
  const float* r = dec + ((size_t)(b * 1024 + 1023)) * DMODEL;
  float s = pb[c];
  for (int k = 0; k < DMODEL; ++k) s += r[k] * pw[(size_t)k * 16 + c];
  last16[tid] = s;
}

__global__ void k_final(const float* __restrict__ last16, const float* __restrict__ ow,
                        const float* __restrict__ ob, float* __restrict__ out)
{
  int t = blockIdx.x * blockDim.x + threadIdx.x;
  if (t >= 16 * 32) return;
  int b = t / 32, p = t % 32;
  float s = ob[p];
  for (int c = 0; c < 16; ++c) s += last16[b * 16 + c] * ow[c * 32 + p];
  out[t] = s;
}

// ---------------------------------------------------------------------------
static inline int sample_cnt(int Lx) {
  int v = 5 * (int)ceil(log((double)Lx));
  return v < Lx ? v : Lx;
}

extern "C" void kernel_launch(void* const* d_in, const int* in_sizes, int n_in,
                              void* d_out, int out_size, void* d_ws, size_t ws_size,
                              hipStream_t stream)
{
  (void)in_sizes; (void)n_in; (void)out_size; (void)ws_size;
  const int B = BB, H = HH;

  // ---- parameters (jax tree_flatten: dict keys sorted) ----
  auto F = [&](int i) { return (const float*)d_in[i]; };
  int pi = 4;
  const float* dec_emb_b = F(pi++); const float* dec_emb_w = F(pi++);
  struct AttnP { const float *Wk, *Wo, *Wq, *Wv, *bk, *bo, *bq, *bv; };
  auto readAttn = [&]() {
    AttnP a;
    a.Wk = F(pi++); a.Wo = F(pi++); a.Wq = F(pi++); a.Wv = F(pi++);
    a.bk = F(pi++); a.bo = F(pi++); a.bq = F(pi++); a.bv = F(pi++);
    return a;
  };
  struct DecL { const float *c1b,*c1w,*c2b,*c2w; AttnP cross;
                const float *ln1b,*ln1g,*ln2b,*ln2g,*ln3b,*ln3g; AttnP self_; };
  DecL dl[2];
  for (int i = 0; i < 2; ++i) {
    dl[i].c1b = F(pi++); dl[i].c1w = F(pi++); dl[i].c2b = F(pi++); dl[i].c2w = F(pi++);
    dl[i].cross = readAttn();
    dl[i].ln1b = F(pi++); dl[i].ln1g = F(pi++); dl[i].ln2b = F(pi++);
    dl[i].ln2g = F(pi++); dl[i].ln3b = F(pi++); dl[i].ln3g = F(pi++);
    dl[i].self_ = readAttn();
  }
  const float* dec_norm_b = F(pi++); const float* dec_norm_g = F(pi++);
  struct ConvP { const float *b, *bnb, *bng, *w; };
  ConvP cv[2];
  for (int i = 0; i < 2; ++i) { cv[i].b = F(pi++); cv[i].bnb = F(pi++); cv[i].bng = F(pi++); cv[i].w = F(pi++); }
  const float* enc_emb_b = F(pi++); const float* enc_emb_w = F(pi++);
  struct EncL { AttnP at; const float *c1b,*c1w,*c2b,*c2w,*ln1b,*ln1g,*ln2b,*ln2g; };
  EncL el[3];
  for (int i = 0; i < 3; ++i) {
    el[i].at = readAttn();
    el[i].c1b = F(pi++); el[i].c1w = F(pi++); el[i].c2b = F(pi++); el[i].c2w = F(pi++);
    el[i].ln1b = F(pi++); el[i].ln1g = F(pi++); el[i].ln2b = F(pi++); el[i].ln2g = F(pi++);
  }
  const float* enc_norm_b = F(pi++); const float* enc_norm_g = F(pi++);
  const float* out_b = F(pi++); const float* out_w = F(pi++);
  const float* proj_b = F(pi++); const float* proj_w = F(pi++);

  // ---- workspace layout ----
  float* wsp = (float*)d_ws;
  size_t off = 0;
  auto alloc = [&](size_t n) { float* p = wsp + off; off += n; return p; };
  const size_t BIG = (size_t)B * 2048 * DMODEL;
  float* bX = alloc(BIG);
  float* bY = alloc(BIG);
  float* bZ = alloc(BIG);
  float* bQ = alloc(BIG);
  float* bK = alloc(BIG);
  float* bV = alloc(BIG);
  float* bC = alloc(BIG);                         // attention context
  float* bS = alloc((size_t)B * H * 40 * 2048);   // sparse scores
  float* bM = alloc((size_t)B * H * 2048);
  int*   bTop = (int*)alloc((size_t)B * H * 40);
  float* bEnc = alloc((size_t)B * 512 * DMODEL);
  float* packw = alloc((size_t)3 * DMODEL * DMODEL);
  float* bnMean = alloc(512);
  float* bnVar  = alloc(512);
  float* vMean  = alloc((size_t)B * H * DH);
  float* last16 = alloc(256);

  auto gemm = [&](const float* A, const float* Bm, const float* bias, float* C,
                  int M, int N, int K, int act, int acc, int shift, int Lseg) {
    dim3 g((N + 127) / 128, (M + 127) / 128);
    k_wmma_gemm<<<g, dim3(256), 0, stream>>>(A, Bm, bias, C, M, N, K, act, acc, shift, Lseg);
  };
  auto addln = [&](const float* x, const float* y, const float* g, const float* b,
                   float* o, int rows) {
    k_add_ln<<<rows, 256, 0, stream>>>(x, y, g, b, o);
  };
  auto prob_attn = [&](const float* q, const float* k, const float* v, float* ctx,
                       float* outBLHD, int Lq, int Lk, int causal, unsigned seed) {
    int U = sample_cnt(Lk);
    int u = sample_cnt(Lq);
    k_sampleM<<<(B * H * Lq + 255) / 256, 256, 0, stream>>>(q, k, bM, Lq, Lk, U, seed);
    k_topk<<<B * H, 256, 0, stream>>>(bM, bTop, Lq, u);
    {
      size_t tot = (size_t)B * H * u * Lk;
      k_scores<<<(unsigned)((tot + 255) / 256), 256, 0, stream>>>(q, k, bTop, bS, Lq, Lk, u, causal);
    }
    k_softmax<<<B * H * u, 256, 0, stream>>>(bS, Lk);
    if (causal) {
      k_ctx_cumsum<<<(B * H * DH + 255) / 256, 256, 0, stream>>>(v, ctx, Lq);
    } else {
      k_vmean<<<(B * H * DH + 255) / 256, 256, 0, stream>>>(v, vMean, Lk);
      size_t tot = (size_t)B * H * Lq * DH;
      k_ctx_fill<<<(unsigned)((tot + 255) / 256), 256, 0, stream>>>(vMean, ctx, Lq);
    }
    k_upd_scatter<<<(B * H * u * DH + 255) / 256, 256, 0, stream>>>(bS, v, bTop, ctx, Lq, Lk, u);
    if (outBLHD) {
      size_t tot = (size_t)B * H * Lq * DH;
      k_mixfalse<<<(unsigned)((tot + 255) / 256), 256, 0, stream>>>(ctx, outBLHD, Lq);
    }
  };

  // ================= Encoder =================
  int L = 2048;
  {
    size_t tot = (size_t)B * L * DMODEL;
    k_embed<<<(unsigned)((tot + 255) / 256), 256, 0, stream>>>(
        (const float*)d_in[0], (const int*)d_in[1], enc_emb_w, enc_emb_b, bX, L);
  }
  for (int li = 0; li < 3; ++li) {
    const EncL& E = el[li];
    int Mr = B * L;
    gemm(bX, E.at.Wq, E.at.bq, bQ, Mr, DMODEL, DMODEL, 0, 0, 0, Mr);
    gemm(bX, E.at.Wk, E.at.bk, bK, Mr, DMODEL, DMODEL, 0, 0, 0, Mr);
    gemm(bX, E.at.Wv, E.at.bv, bV, Mr, DMODEL, DMODEL, 0, 0, 0, Mr);
    prob_attn(bQ, bK, bV, bC, bY, L, L, 0, 0x9E3779B9u + li);
    gemm(bY, E.at.Wo, E.at.bo, bZ, Mr, DMODEL, DMODEL, 0, 0, 0, Mr);
    addln(bX, bZ, E.ln1g, E.ln1b, bY, Mr);
    gemm(bY, E.c1w, E.c1b, bZ, Mr, DMODEL, DMODEL, 1, 0, 0, Mr);   // GELU
    gemm(bZ, E.c2w, E.c2b, bQ, Mr, DMODEL, DMODEL, 0, 0, 0, Mr);
    addln(bY, bQ, E.ln2g, E.ln2b, bX, Mr);
    if (li < 2) {
      const ConvP& C = cv[li];
      k_packw<<<(3 * DMODEL * DMODEL + 255) / 256, 256, 0, stream>>>(C.w, packw);
      gemm(bX, packw,                         C.b,     bY, Mr, DMODEL, DMODEL, 0, 0, -1, L);
      gemm(bX, packw + (size_t)DMODEL*DMODEL, nullptr, bY, Mr, DMODEL, DMODEL, 0, 1,  0, L);
      gemm(bX, packw + (size_t)2*DMODEL*DMODEL, nullptr, bY, Mr, DMODEL, DMODEL, 0, 1,  1, L);
      k_bnstats<<<DMODEL, 256, 0, stream>>>(bY, bnMean, bnVar, Mr);
      int Lo = L / 2;
      size_t tot = (size_t)B * Lo * DMODEL;
      k_bnpool<<<(unsigned)((tot + 255) / 256), 256, 0, stream>>>(
          bY, bnMean, bnVar, C.bng, C.bnb, bX, L, Lo);
      L = Lo;
    }
  }
  addln(bX, nullptr, enc_norm_g, enc_norm_b, bEnc, B * L);
  const int Lenc = L;   // 512

  // ================= Decoder =================
  const int Ld = 1024;
  const int Md = B * Ld;
  {
    size_t tot = (size_t)B * Ld * DMODEL;
    k_embed<<<(unsigned)((tot + 255) / 256), 256, 0, stream>>>(
        (const float*)d_in[2], (const int*)d_in[3], dec_emb_w, dec_emb_b, bX, Ld);
  }
  for (int li = 0; li < 2; ++li) {
    const DecL& D = dl[li];
    // causal self ProbSparse attention (mix=True -> ctx (B,H,L,DH) fed directly)
    gemm(bX, D.self_.Wq, D.self_.bq, bQ, Md, DMODEL, DMODEL, 0, 0, 0, Md);
    gemm(bX, D.self_.Wk, D.self_.bk, bK, Md, DMODEL, DMODEL, 0, 0, 0, Md);
    gemm(bX, D.self_.Wv, D.self_.bv, bV, Md, DMODEL, DMODEL, 0, 0, 0, Md);
    prob_attn(bQ, bK, bV, bC, nullptr, Ld, Ld, 1, 0xB5297A4Du + li);
    gemm(bC, D.self_.Wo, D.self_.bo, bZ, Md, DMODEL, DMODEL, 0, 0, 0, Md);
    addln(bX, bZ, D.ln1g, D.ln1b, bY, Md);
    // cross full attention against encoder output
    gemm(bY,   D.cross.Wq, D.cross.bq, bQ, Md,       DMODEL, DMODEL, 0, 0, 0, Md);
    gemm(bEnc, D.cross.Wk, D.cross.bk, bK, B * Lenc, DMODEL, DMODEL, 0, 0, 0, B * Lenc);
    gemm(bEnc, D.cross.Wv, D.cross.bv, bV, B * Lenc, DMODEL, DMODEL, 0, 0, 0, B * Lenc);
    k_fullattn<<<B * H * Ld, 128, 0, stream>>>(bQ, bK, bV, bC, Ld, Lenc);
    gemm(bC, D.cross.Wo, D.cross.bo, bZ, Md, DMODEL, DMODEL, 0, 0, 0, Md);
    addln(bY, bZ, D.ln2g, D.ln2b, bX, Md);
    // FFN
    gemm(bX, D.c1w, D.c1b, bZ, Md, DMODEL, DMODEL, 1, 0, 0, Md);
    gemm(bZ, D.c2w, D.c2b, bQ, Md, DMODEL, DMODEL, 0, 0, 0, Md);
    addln(bX, bQ, D.ln3g, D.ln3b, bY, Md);
    { float* t = bX; bX = bY; bY = t; }
  }
  addln(bX, nullptr, dec_norm_g, dec_norm_b, bY, Md);

  k_lastproj<<<1, 256, 0, stream>>>(bY, proj_w, proj_b, last16);
  k_final<<<2, 256, 0, stream>>>(last16, out_w, out_b, (float*)d_out);
}
